// BasicBlock_37228776522121
// MI455X (gfx1250) — 8-bit matrix operand order repaired and checked by reading, compiled, not run on hardware
//
#include <hip/hip_runtime.h>

typedef __attribute__((ext_vector_type(8))) int v8i;

#define C_IN   256
#define H_DIM  56
#define W_DIM  56
#define HW     3136
#define N_B    64
#define COUT   256
#define K_TOT  2304
#define M_TOT  (N_B * HW)

#define WS_WPACK  0
#define WS_SCALE  589824
#define WS_BIAS   590848
#define WS_XBIN   591872

#define LDS_STRIDE 272

__global__ __launch_bounds__(256)
void bb_binarize_x(const float* __restrict__ x, signed char* __restrict__ xbin) {
    long g = (long)blockIdx.x * blockDim.x + threadIdx.x;
    int  c4   = (int)(g & 63);
    long rest = g >> 6;
    int  hw   = (int)(rest % HW);
    int  n    = (int)(rest / HW);
    int  c0   = c4 * 4;
    const float* xp = x + (long)n * C_IN * HW + hw;
    unsigned packed = 0;
    #pragma unroll
    for (int j = 0; j < 4; ++j) {
        float v = xp[(long)(c0 + j) * HW];
        unsigned b = (v >= 0.0f) ? 0x01u : 0xFFu;
        packed |= b << (8 * j);
    }
    ((unsigned*)xbin)[g] = packed;
}

__global__ __launch_bounds__(256)
void bb_pack_w(const float* __restrict__ W, const float* __restrict__ gamma,
               const float* __restrict__ beta, const float* __restrict__ rmean,
               const float* __restrict__ rvar,
               signed char* __restrict__ wpack, float* __restrict__ scale,
               float* __restrict__ bias) {
    int co = blockIdx.x;
    int c  = threadIdx.x;
    const float* wp = W + ((long)co * C_IN + c) * 9;
    float s = 0.0f;
    signed char sg[9];
    #pragma unroll
    for (int j = 0; j < 9; ++j) {
        float v = wp[j];
        s += fabsf(v);
        sg[j] = (v >= 0.0f) ? (signed char)1 : (signed char)-1;
    }
    __shared__ float red[256];
    red[c] = s;
    __syncthreads();
    for (int off = 128; off > 0; off >>= 1) {
        if (c < off) red[c] += red[c + off];
        __syncthreads();
    }
    #pragma unroll
    for (int j = 0; j < 9; ++j)
        wpack[(long)co * K_TOT + j * C_IN + c] = sg[j];
    if (c == 0) {
        float alpha = red[0] * (1.0f / 2304.0f);
        float inv   = gamma[co] * rsqrtf(rvar[co] + 1e-5f);
        scale[co]   = alpha * inv;
        bias[co]    = beta[co] - rmean[co] * inv;
    }
}

__device__ __forceinline__ void async_stage64(signed char* ldsdst,
                                              const signed char* gsrc) {
    unsigned lds_off = (unsigned)(size_t)ldsdst;
    unsigned long long ga = (unsigned long long)(size_t)gsrc;
    asm volatile("global_load_async_to_lds_b128 %0, %1, off offset:0"
                 :: "v"(lds_off), "v"(ga) : "memory");
    asm volatile("global_load_async_to_lds_b128 %0, %1, off offset:16"
                 :: "v"(lds_off), "v"(ga) : "memory");
    asm volatile("global_load_async_to_lds_b128 %0, %1, off offset:32"
                 :: "v"(lds_off), "v"(ga) : "memory");
    asm volatile("global_load_async_to_lds_b128 %0, %1, off offset:48"
                 :: "v"(lds_off), "v"(ga) : "memory");
}

__device__ __forceinline__ void wait_async0() {
    asm volatile("s_wait_asynccnt 0x0" ::: "memory");
}

__device__ __forceinline__ v8i load_a_frag(const signed char* ar, int khalf) {
    int av[8];
    #pragma unroll
    for (int p = 0; p < 4; ++p) {
        unsigned long long d =
            *(const unsigned long long*)(ar + (p >> 1) * 32 + khalf * 16 + (p & 1) * 8);
        av[2 * p + 0] = (int)(unsigned)(d & 0xffffffffull);
        av[2 * p + 1] = (int)(d >> 32);
    }
    v8i a = {av[0], av[1], av[2], av[3], av[4], av[5], av[6], av[7]};
    return a;
}

__device__ __forceinline__ v8i load_b_frag(const signed char* wb, int kbg, int khalf) {
    int4 b0 = *(const int4*)(wb + kbg + khalf * 16);
    int4 b1 = *(const int4*)(wb + kbg + 32 + khalf * 16);
    v8i b = {b0.x, b0.y, b0.z, b0.w, b1.x, b1.y, b1.z, b1.w};
    return b;
}

__global__ __launch_bounds__(256)
void bb_conv_wmma(const signed char* __restrict__ xbin,
                  const signed char* __restrict__ wpack,
                  const float* __restrict__ scale, const float* __restrict__ bias,
                  const float* __restrict__ x, float* __restrict__ out) {
    __shared__ signed char Atile[2][64 * LDS_STRIDE];

    const int tid  = threadIdx.x;
    const int lane = tid & 31;
    const int wave = tid >> 5;
    const int mgrp = wave & 1;
    const int ngrp = wave >> 1;

    const int pbase = blockIdx.x * 64;
    const int cbase = blockIdx.y * 128;

    const int sm  = tid >> 2;
    const int sb0 = (tid & 3) * 64;
    const int sp  = pbase + sm;
    const int sn  = sp / HW;
    const int shw = sp % HW;
    const int sh  = shw / W_DIM;
    const int sw  = shw % W_DIM;
    const long srcPix = (long)sn * HW;

    const int khalf = lane >> 4;
    const int l15   = lane & 15;
    const signed char* wb0 = wpack + (long)(cbase + ngrp * 32 + l15) * K_TOT;
    const signed char* wb1 = wb0 + 16 * (long)K_TOT;

    v8i acc[2][2];
    #pragma unroll
    for (int mi = 0; mi < 2; ++mi)
        #pragma unroll
        for (int ni = 0; ni < 2; ++ni) {
            v8i z = {0, 0, 0, 0, 0, 0, 0, 0};
            acc[mi][ni] = z;
        }

    auto stage_tap = [&](int rs, int buf) {
        const int r   = rs / 3;
        const int s2  = rs % 3;
        const int hin = sh + r - 1;
        const int win = sw + s2 - 1;
        signed char* dst = &Atile[buf][sm * LDS_STRIDE + sb0];
        if (((unsigned)hin < H_DIM) && ((unsigned)win < W_DIM)) {
            const signed char* src =
                xbin + (srcPix + (long)hin * W_DIM + win) * C_IN + sb0;
            async_stage64(dst, src);
        } else {
            int4 z = {0, 0, 0, 0};
            #pragma unroll
            for (int j = 0; j < 4; ++j)
                *(int4*)(dst + j * 16) = z;
        }
    };

    stage_tap(0, 0);

    for (int rs = 0; rs < 9; ++rs) {
        wait_async0();
        __syncthreads();

        if (rs < 8) stage_tap(rs + 1, (rs + 1) & 1);

        const signed char* ab  = Atile[rs & 1];
        const signed char* ar0 = ab + (mgrp * 32 + l15) * LDS_STRIDE;
        const signed char* ar1 = ar0 + 16 * LDS_STRIDE;

        #pragma unroll
        for (int cb = 0; cb < 4; ++cb) {
            const int co  = cb * 64;
            const int kbg = rs * 256 + co;

            v8i a0 = load_a_frag(ar0 + co, khalf);
            v8i a1 = load_a_frag(ar1 + co, khalf);
            v8i b0 = load_b_frag(wb0, kbg, khalf);
            v8i b1 = load_b_frag(wb1, kbg, khalf);

            acc[0][0] = __builtin_amdgcn_wmma_i32_16x16x64_iu8(
                true, a0, true, b0, acc[0][0], false, false);
            acc[0][1] = __builtin_amdgcn_wmma_i32_16x16x64_iu8(
                true, a0, true, b1, acc[0][1], false, false);
            acc[1][0] = __builtin_amdgcn_wmma_i32_16x16x64_iu8(
                true, a1, true, b0, acc[1][0], false, false);
            acc[1][1] = __builtin_amdgcn_wmma_i32_16x16x64_iu8(
                true, a1, true, b1, acc[1][1], false, false);
        }
    }

    const int ch0   = cbase + ngrp * 32 + l15;
    const int prow0 = pbase + mgrp * 32 + 8 * khalf;
    #pragma unroll
    for (int mi = 0; mi < 2; ++mi) {
        const int p  = prow0 + mi * 16;
        const int n  = p / HW;
        const int hw = p % HW;
        #pragma unroll
        for (int ni = 0; ni < 2; ++ni) {
            const int ch = ch0 + ni * 16;
            const float sc = scale[ch];
            const float bi = bias[ch];
            const long oi = ((long)(n * COUT + ch)) * HW + hw;
            v8i a = acc[mi][ni];
            float4 r0 = *(const float4*)(x + oi);
            float4 r1 = *(const float4*)(x + oi + 4);
            float4 o0 = {(float)a[0] * sc + bi + r0.x,
                         (float)a[1] * sc + bi + r0.y,
                         (float)a[2] * sc + bi + r0.z,
                         (float)a[3] * sc + bi + r0.w};
            float4 o1 = {(float)a[4] * sc + bi + r1.x,
                         (float)a[5] * sc + bi + r1.y,
                         (float)a[6] * sc + bi + r1.z,
                         (float)a[7] * sc + bi + r1.w};
            *(float4*)(out + oi)     = o0;
            *(float4*)(out + oi + 4) = o1;
        }
    }
}

extern "C" void kernel_launch(void* const* d_in, const int* in_sizes, int n_in,
                              void* d_out, int out_size, void* d_ws, size_t ws_size,
                              hipStream_t stream) {
    const float* x     = (const float*)d_in[0];
    const float* W     = (const float*)d_in[1];
    const float* gamma = (const float*)d_in[2];
    const float* beta  = (const float*)d_in[3];
    const float* rmean = (const float*)d_in[4];
    const float* rvar  = (const float*)d_in[5];
    float* out = (float*)d_out;

    char* ws = (char*)d_ws;
    signed char* wpack = (signed char*)(ws + WS_WPACK);
    float*       scale = (float*)(ws + WS_SCALE);
    float*       bias  = (float*)(ws + WS_BIAS);
    signed char* xbin  = (signed char*)(ws + WS_XBIN);

    long groups = (long)N_B * HW * C_IN / 4;
    bb_binarize_x<<<(int)(groups / 256), 256, 0, stream>>>(x, xbin);

    bb_pack_w<<<COUT, 256, 0, stream>>>(W, gamma, beta, rmean, rvar,
                                        wpack, scale, bias);

    dim3 grid(M_TOT / 64, COUT / 128);
    bb_conv_wmma<<<grid, 256, 0, stream>>>(xbin, wpack, scale, bias, x, out);
}
